// SpatiotemporalAttentionCrossInteractionFuse_4020089389353
// MI455X (gfx1250) — matrix operand order repaired and checked by reading, not compiled, not run on hardware
//
#include <hip/hip_runtime.h>
#include <math.h>

// ---------------- problem constants ----------------
#define Tn  256      // channels / frames
#define Vn  90       // joints
#define Vp  96       // padded joints (6 x 16 WMMA tiles)
#define CIn 128      // inter channels
#define NB  1024     // batch
#define TV  (Tn * Vn)   // 23040

typedef __bf16 bf16;
typedef __attribute__((ext_vector_type(16))) __bf16 v16bf;
typedef __attribute__((ext_vector_type(8)))  float  v8f;
typedef __attribute__((ext_vector_type(4)))  unsigned int u32x4;   // POD 16B vector (HIP uint4 is a class)
typedef __attribute__((ext_vector_type(4)))  float  f32x4;

// explicit address spaces so DS/global paths stay inferred (no generic flat fallback)
#define AS3 __attribute__((address_space(3)))
#define AS1 __attribute__((address_space(1)))
typedef AS3 bf16          lbf16;
typedef AS3 float         lfloat;
typedef AS3 char          lchar;
typedef AS3 unsigned int  luint;
typedef const AS1 bf16    gbf16;
typedef const AS1 float   gf32;

// ---------------- LDS layout (element strides, byte offsets) ----------------
#define XT_LD 272   // [Vp][Tn] bf16 transposed inputs (pad cols 256..271 never read)
#define RM_LD 104   // [CIn][Vp] bf16 row-major
#define TR_LD 136   // [Vp][CIn] bf16 transposed
#define ET_LD 132   // [128][128] f32
#define ES_LD 100   // [96][96]  f32
#define AT_LD 136   // [128][128] bf16 attn-time
#define AS_LD 104   // [96][96]  bf16 attn-space (transposed)
#define PR_LD 104   // [Tn][Vp]  bf16 proj

// region B (0 .. 161792)
#define G1T_OFF  0        // 96*136*2 = 26112
#define G2T_OFF  26112
#define TH_OFF   52224    // 128*104*2 = 26624
#define PH_OFF   78848
#define THT_OFF  105472
#define PHT_OFF  131584   // end 157696
#define AT1_OFF  52224    // reuses TH.. after E GEMMs (34816 each)
#define AT2_OFF  87040
#define AS1T_OFF 121856   // 19968 each
#define AS2T_OFF 141824   // end 161792
#define P1_OFF   52224    // reuses attn area after Z GEMMs (53248 each)
#define P2_OFF   105472   // end 158720
// scratch region (161792 .. 267776) — phase overlaid
#define SC_OFF   161792
#define XT1_OFF  (SC_OFF)           // 52224
#define XT2_OFF  (SC_OFF + 52224)
#define ET_OFF   (SC_OFF)           // 67584
#define ES_OFF   (SC_OFF + 67584)   // 38400 -> end SC+105984
#define Y1_OFF   (SC_OFF)           // 26624
#define Y2_OFF   (SC_OFF + 26624)
#define Z1T_OFF  (SC_OFF + 53248)   // 26112
#define Z2T_OFF  (SC_OFF + 79360)
// small region
#define X1MV_OFF 267776   // 256 f32
#define X2MT_OFF 268800   // 96 f32 (padded)
#define SCL_OFF  269184   // 32 f32
#define LDS_BYTES 269312

// ---------------- workspace layout (bytes) ----------------
#define WS_GW   0        // 128*256 bf16 = 65536
#define WS_THW  65536
#define WS_PHW  131072
#define WS_WW   196608   // 256*128 bf16
#define WS_GB   262144   // 128 f32
#define WS_THB  262656
#define WS_PHB  263168
#define WS_SW   263680   // 256 f32
#define WS_OW   264704   // 256 f32

// ---------------- typed loads per address space ----------------
static __device__ __forceinline__ u32x4 ld16b(const lbf16* p) { return *(const AS3 u32x4*)p; }
static __device__ __forceinline__ u32x4 ld16b(gbf16* p)       { return *(const AS1 u32x4*)p; }
static __device__ __forceinline__ f32x4 ld4f(gf32* p)         { return *(const AS1 f32x4*)p; }

// A fragment: 16x32 bf16, row-major source, stride ld (elements).
// Lane L: m = L&15 ; chunk0 @ k0 + 8*(L>>4), chunk1 = chunk0 + 16.  (ISA 7.12.2)
template <typename P>
static __device__ __forceinline__ v16bf afrag(P base, int ld, int r0, int k0) {
  const int lane = threadIdx.x & 31;
  const int m = lane & 15, hi = lane >> 4;
  P p = base + (r0 + m) * ld + (k0 + hi * 8);
  union { u32x4 u; __bf16 h[8]; } a0, a1;
  a0.u = ld16b(p);
  a1.u = ld16b(p + 16);
  v16bf f;
#pragma unroll
  for (int j = 0; j < 8; ++j) { f[j] = a0.h[j]; f[8 + j] = a1.h[j]; }
  return f;
}

// B fragment: 32x16 bf16 from a pre-TRANSPOSED LDS buffer Bt[n][k] (k contiguous).
// Lane L: n = L&15 ; chunk0 @ k0 + 8*(L>>4), chunk1 = chunk0 + 16 (matches afrag).
static __device__ __forceinline__ v16bf bfragT(const lbf16* baseT, int ld, int n0, int k0) {
  const int lane = threadIdx.x & 31;
  const int n = lane & 15, hi = lane >> 4;
  const lbf16* p = baseT + (n0 + n) * ld + (k0 + hi * 8);
  union { u32x4 u; __bf16 h[8]; } b0, b1;
  b0.u = ld16b(p);
  b1.u = ld16b(p + 16);
  v16bf f;
#pragma unroll
  for (int j = 0; j < 8; ++j) { f[j] = b0.h[j]; f[8 + j] = b1.h[j]; }
  return f;
}

// packed pair store into LDS: two contiguous bf16 as one 32-bit DS store
static __device__ __forceinline__ void st2l(lbf16* p, float lo, float hi) {
  union { unsigned int u; __bf16 h[2]; } t;
  t.h[0] = (bf16)lo; t.h[1] = (bf16)hi;
  *(luint*)p = t.u;
}

#define WMMA_BF16(acc, a, b) \
  __builtin_amdgcn_wmma_f32_16x16x32_bf16(false, (a), false, (b), (short)0, (acc), false, false)

// Projection GEMM: C(128x96) = W_eff(128x256) x X(256x96), X from transposed LDS.
// Software-pipelined: next K-step fragments are fetched before the current WMMA.
static __device__ void proj_gemm(gbf16* W, gf32* bias,
                                 const lbf16* Xt, lbf16* outRM, lbf16* outTR, const bool hasRM) {
  const int wave = threadIdx.x >> 5;
  const int lane = threadIdx.x & 31;
  const int nl = lane & 15, hi = lane >> 4;
  for (int tile = wave; tile < 48; tile += 8) {
    const int tm = tile / 6, tn = tile % 6;
    // bias[tm*16 + 8*hi .. +7] for this lane's 8 output rows: two unconditional 16B loads
    const f32x4 bL = ld4f(bias + tm * 16 + 8 * hi);
    const f32x4 bH = ld4f(bias + tm * 16 + 8 * hi + 4);
    v16bf a = afrag(W, Tn, tm * 16, 0);
    v16bf b = bfragT(Xt, XT_LD, tn * 16, 0);
    v8f acc = {};
#pragma unroll
    for (int k0 = 0; k0 < Tn; k0 += 32) {
      const v16bf ca = a, cb = b;
      if (k0 + 32 < Tn) {
        a = afrag(W, Tn, tm * 16, k0 + 32);
        b = bfragT(Xt, XT_LD, tn * 16, k0 + 32);
      }
      acc = WMMA_BF16(acc, ca, cb);
    }
    const int n = tn * 16 + nl;
    const bool valid = (n < Vn);     // zero padded joint columns (incl. bias!)
#pragma unroll
    for (int r = 0; r < 8; r += 2) {
      const int m = tm * 16 + r + 8 * hi;
      const float br0 = (r < 4) ? bL[r & 3]       : bH[r & 3];
      const float br1 = (r < 4) ? bL[(r + 1) & 3] : bH[(r + 1) & 3];
      const float v0 = valid ? (acc[r]     + br0) : 0.f;
      const float v1 = valid ? (acc[r + 1] + br1) : 0.f;
      if (hasRM) { outRM[m * RM_LD + n] = (bf16)v0; outRM[(m + 1) * RM_LD + n] = (bf16)v1; }
      st2l(outTR + n * TR_LD + m, v0, v1);
    }
  }
}

// ---------------- prep kernels: fold BN into bf16 weights ----------------
__global__ void prep_w_kernel(const float* __restrict__ gw_f, const float* __restrict__ g_bng, const float* __restrict__ g_bnv,
                              const float* __restrict__ thw_f, const float* __restrict__ th_bng, const float* __restrict__ th_bnv,
                              const float* __restrict__ phw_f, const float* __restrict__ ph_bng, const float* __restrict__ ph_bnv,
                              const float* __restrict__ Ww_f,
                              bf16* gw_o, bf16* thw_o, bf16* phw_o, bf16* ww_o) {
  const int idx = blockIdx.x * 256 + threadIdx.x;   // 4*32768 total
  const int which = idx >> 15;
  const int r = idx & 32767;
  if (which == 3) { ww_o[r] = (bf16)Ww_f[r]; return; }
  const int t = r & 255;
  const float* w  = (which == 0) ? gw_f  : (which == 1) ? thw_f  : phw_f;
  const float* bg = (which == 0) ? g_bng : (which == 1) ? th_bng : ph_bng;
  const float* bv = (which == 0) ? g_bnv : (which == 1) ? th_bnv : ph_bnv;
  bf16* o         = (which == 0) ? gw_o  : (which == 1) ? thw_o  : phw_o;
  const float scale = bg[t] * rsqrtf(bv[t] + 1e-5f);
  o[r] = (bf16)(w[r] * scale);
}

__global__ void prep_b_kernel(const float* gw_f, const float* g_bng, const float* g_bnb, const float* g_bnm, const float* g_bnv, const float* g_b,
                              const float* thw_f, const float* th_bng, const float* th_bnb, const float* th_bnm, const float* th_bnv, const float* th_b,
                              const float* phw_f, const float* ph_bng, const float* ph_bnb, const float* ph_bnm, const float* ph_bnv, const float* ph_b,
                              const float* W_b, const float* W_bng, const float* W_bnb, const float* W_bnm, const float* W_bnv,
                              float* gb_o, float* thb_o, float* phb_o, float* sW_o, float* oW_o) {
  const int tid = threadIdx.x;  // 256 threads
  if (tid < CIn) {
    const float* W3[3] = {gw_f, thw_f, phw_f};
    const float* BG[3] = {g_bng, th_bng, ph_bng};
    const float* BB[3] = {g_bnb, th_bnb, ph_bnb};
    const float* BM[3] = {g_bnm, th_bnm, ph_bnm};
    const float* BV[3] = {g_bnv, th_bnv, ph_bnv};
    const float* Bb[3] = {g_b, th_b, ph_b};
    float* Ob[3] = {gb_o, thb_o, phb_o};
    for (int q = 0; q < 3; ++q) {
      float s = 0.f;
      for (int t = 0; t < Tn; ++t) {
        const float sc = BG[q][t] * rsqrtf(BV[q][t] + 1e-5f);
        s += W3[q][tid * Tn + t] * (BB[q][t] - BM[q][t] * sc);
      }
      Ob[q][tid] = Bb[q][tid] + s;
    }
  }
  // post-BN for W branch, folded with W bias: proj = raw*sW + oW
  const float s = W_bng[tid] * rsqrtf(W_bnv[tid] + 1e-5f);
  sW_o[tid] = s;
  oW_o[tid] = W_b[tid] * s + (W_bnb[tid] - W_bnm[tid] * s);
}

// ---------------- fused per-batch kernel ----------------
__global__ __launch_bounds__(256)
void fused_attn_kernel(const float* __restrict__ x1, const float* __restrict__ x2,
                       const bf16* __restrict__ gw_, const bf16* __restrict__ thw_,
                       const bf16* __restrict__ phw_, const bf16* __restrict__ ww_,
                       const float* __restrict__ gb_, const float* __restrict__ thb_,
                       const float* __restrict__ phb_, const float* __restrict__ sW_,
                       const float* __restrict__ oW_, float* __restrict__ out) {
  extern __shared__ char smem[];
  lchar* sm = (lchar*)smem;                 // AS3 base; generic<->AS3 casts fold
  gbf16* gw  = (gbf16*)gw_;
  gbf16* thw = (gbf16*)thw_;
  gbf16* phw = (gbf16*)phw_;
  gbf16* ww  = (gbf16*)ww_;
  gf32* gb  = (gf32*)gb_;
  gf32* thb = (gf32*)thb_;
  gf32* phb = (gf32*)phb_;
  gf32* sW  = (gf32*)sW_;
  gf32* oW  = (gf32*)oW_;

  const int tid  = threadIdx.x;
  const int wave = tid >> 5;
  const int lane = tid & 31;
  const int nl = lane & 15, hi = lane >> 4;
  const size_t gbase = (size_t)blockIdx.x * TV;

  lbf16* G1t = (lbf16*)(sm + G1T_OFF);
  lbf16* G2t = (lbf16*)(sm + G2T_OFF);
  lbf16* TH  = (lbf16*)(sm + TH_OFF);
  lbf16* PH  = (lbf16*)(sm + PH_OFF);
  lbf16* THt = (lbf16*)(sm + THT_OFF);
  lbf16* PHt = (lbf16*)(sm + PHT_OFF);
  lbf16* at1 = (lbf16*)(sm + AT1_OFF);
  lbf16* at2 = (lbf16*)(sm + AT2_OFF);
  lbf16* as1t = (lbf16*)(sm + AS1T_OFF);
  lbf16* as2t = (lbf16*)(sm + AS2T_OFF);
  lbf16* P1  = (lbf16*)(sm + P1_OFF);
  lbf16* P2  = (lbf16*)(sm + P2_OFF);
  lbf16* Xt1 = (lbf16*)(sm + XT1_OFF);
  lbf16* Xt2 = (lbf16*)(sm + XT2_OFF);
  lfloat* Et = (lfloat*)(sm + ET_OFF);
  lfloat* Es = (lfloat*)(sm + ES_OFF);
  lbf16* Y1  = (lbf16*)(sm + Y1_OFF);
  lbf16* Y2  = (lbf16*)(sm + Y2_OFF);
  lbf16* Z1t = (lbf16*)(sm + Z1T_OFF);
  lbf16* Z2t = (lbf16*)(sm + Z2T_OFF);
  lfloat* x1mv = (lfloat*)(sm + X1MV_OFF);
  lfloat* x2mt = (lfloat*)(sm + X2MT_OFF);
  lfloat* scl  = (lfloat*)(sm + SCL_OFF);

  // ---- Phase A: stage transposed bf16 inputs, zero pads ----
  if (tid < 32) scl[tid] = 0.f;
  {
    luint* z1 = (luint*)(Xt1 + Vn * XT_LD);
    luint* z2 = (luint*)(Xt2 + Vn * XT_LD);
    for (int i = tid; i < (6 * XT_LD) / 2; i += 256) { z1[i] = 0u; z2[i] = 0u; }
  }
  {
    int t = tid / Vn, v = tid - (tid / Vn) * Vn;    // i = tid; i += 256 == (t+=2, v+=76) mod 90
    for (int i = tid; i < TV; i += 256) {
      Xt1[v * XT_LD + t] = (bf16)x1[gbase + i];
      Xt2[v * XT_LD + t] = (bf16)x2[gbase + i];
      v += 76; t += 2; if (v >= Vn) { v -= Vn; t += 1; }
    }
  }
  __syncthreads();

  // ---- Phase B: per-row/col means (cross terms; singleton softmax == 1) ----
  {
    float s = 0.f;
    const float* p = x1 + gbase + (size_t)tid * Vn;
    for (int v = 0; v < Vn; ++v) s += p[v];
    x1mv[tid] = s;                       // sum over joints, row t = tid
  }
  if (tid < Vp) {
    float s = 0.f;
    if (tid < Vn) for (int t = 0; t < Tn; ++t) s += (float)Xt2[tid * XT_LD + t];
    x2mt[tid] = s;                       // sum over frames, joint v = tid
  }
  __syncthreads();
  if (tid == 0) { float s = 0; for (int t = 0; t < Tn; ++t) s += x1mv[t]; scl[0] = s; }
  if (tid == 1) { float s = 0; for (int v = 0; v < Vn; ++v) s += x2mt[v]; scl[1] = s; }

  // ---- Phase 1: BN-folded projections (4 GEMMs, 128x96, K=256) ----
  proj_gemm(thw, thb, Xt1, TH, THt, true);
  proj_gemm(phw, phb, Xt2, PH, PHt, true);
  proj_gemm(gw,  gb,  Xt1, TH /*unused*/, G1t, false);
  proj_gemm(gw,  gb,  Xt2, TH /*unused*/, G2t, false);
  __syncthreads();

  // ---- Phase 2: energies.  Et = TH·PHᵀ (K=Vp), Es = THᵀ·PH (K=CIn) ----
  for (int tile = wave; tile < 64; tile += 8) {
    const int tm = tile >> 3, tn = tile & 7;
    v16bf a = afrag(TH, RM_LD, tm * 16, 0);
    v16bf b = bfragT(PH, RM_LD, tn * 16, 0);
    v8f acc = {};
#pragma unroll
    for (int k0 = 0; k0 < Vp; k0 += 32) {
      const v16bf ca = a, cb = b;
      if (k0 + 32 < Vp) {
        a = afrag(TH, RM_LD, tm * 16, k0 + 32);
        b = bfragT(PH, RM_LD, tn * 16, k0 + 32);
      }
      acc = WMMA_BF16(acc, ca, cb);
    }
#pragma unroll
    for (int r = 0; r < 8; ++r)
      Et[(tm * 16 + r + 8 * hi) * ET_LD + tn * 16 + nl] = acc[r];
  }
  for (int tile = wave; tile < 36; tile += 8) {
    const int tm = tile / 6, tn = tile % 6;
    v16bf a = afrag(THt, TR_LD, tm * 16, 0);
    v16bf b = bfragT(PHt, TR_LD, tn * 16, 0);
    v8f acc = {};
#pragma unroll
    for (int k0 = 0; k0 < CIn; k0 += 32) {
      const v16bf ca = a, cb = b;
      if (k0 + 32 < CIn) {
        a = afrag(THt, TR_LD, tm * 16, k0 + 32);
        b = bfragT(PHt, TR_LD, tn * 16, k0 + 32);
      }
      acc = WMMA_BF16(acc, ca, cb);
    }
#pragma unroll
    for (int r = 0; r < 8; ++r)
      Es[(tm * 16 + r + 8 * hi) * ES_LD + tn * 16 + nl] = acc[r];
  }
  __syncthreads();

  // ---- Phase 3: four softmaxes, written in GEMM-ready layouts ----
  if (tid < CIn) {
    const int c = tid;
    float rmax = -1e30f, cmax = -1e30f;
    for (int d = 0; d < CIn; ++d) {
      rmax = fmaxf(rmax, Et[c * ET_LD + d]);
      cmax = fmaxf(cmax, Et[d * ET_LD + c]);
    }
    float rs = 0.f, cs = 0.f;
    for (int d = 0; d < CIn; ++d) {
      rs += __expf(Et[c * ET_LD + d] - rmax);
      cs += __expf(Et[d * ET_LD + c] - cmax);
    }
    const float ri = 1.f / rs, ci = 1.f / cs;
    for (int d = 0; d < CIn; d += 2) {  // packed pair stores (contiguous in d)
      st2l(at1 + c * AT_LD + d,
           __expf(Et[c * ET_LD + d] - rmax) * ri,
           __expf(Et[c * ET_LD + d + 1] - rmax) * ri);          // softmax(Et)
      st2l(at2 + c * AT_LD + d,
           __expf(Et[d * ET_LD + c] - cmax) * ci,
           __expf(Et[(d + 1) * ET_LD + c] - cmax) * ci);        // softmax(Etᵀ)
    }
  } else if (tid < CIn + Vp) {
    const int v = tid - CIn;
    if (v < Vn) {
      float rmax = -1e30f, cmax = -1e30f;
      for (int w = 0; w < Vn; ++w) {
        rmax = fmaxf(rmax, Es[v * ES_LD + w]);
        cmax = fmaxf(cmax, Es[w * ES_LD + v]);
      }
      float rs = 0.f, cs = 0.f;
      for (int w = 0; w < Vn; ++w) {
        rs += __expf(Es[v * ES_LD + w] - rmax);
        cs += __expf(Es[w * ES_LD + v] - cmax);
      }
      const float ri = 1.f / rs, ci = 1.f / cs;
      for (int w = 0; w < Vp; ++w) {
        as2t[w * AS_LD + v] = (bf16)((w < Vn) ? __expf(Es[v * ES_LD + w] - rmax) * ri : 0.f);
        as1t[w * AS_LD + v] = (bf16)((w < Vn) ? __expf(Es[w * ES_LD + v] - cmax) * ci : 0.f);
      }
    } else {
      for (int w = 0; w < Vp; ++w) { as2t[w * AS_LD + v] = (bf16)0.f; as1t[w * AS_LD + v] = (bf16)0.f; }
    }
  }
  __syncthreads();

  // ---- Phase 4: Y1 = at2·G1, Y2 = at1·G2   (128x96, K=128) ----
  for (int tile = wave; tile < 48; tile += 8) {
    const int tm = tile / 6, tn = tile % 6;
    v16bf a1 = afrag(at2, AT_LD, tm * 16, 0);
    v16bf a2 = afrag(at1, AT_LD, tm * 16, 0);
    v16bf b1 = bfragT(G1t, TR_LD, tn * 16, 0);
    v16bf b2 = bfragT(G2t, TR_LD, tn * 16, 0);
    v8f acc1 = {}, acc2 = {};
#pragma unroll
    for (int k0 = 0; k0 < CIn; k0 += 32) {
      const v16bf ca1 = a1, ca2 = a2, cb1 = b1, cb2 = b2;
      if (k0 + 32 < CIn) {
        a1 = afrag(at2, AT_LD, tm * 16, k0 + 32);
        a2 = afrag(at1, AT_LD, tm * 16, k0 + 32);
        b1 = bfragT(G1t, TR_LD, tn * 16, k0 + 32);
        b2 = bfragT(G2t, TR_LD, tn * 16, k0 + 32);
      }
      acc1 = WMMA_BF16(acc1, ca1, cb1);
      acc2 = WMMA_BF16(acc2, ca2, cb2);
    }
    const int n = tn * 16 + nl;
#pragma unroll
    for (int r = 0; r < 8; ++r) {
      const int m = tm * 16 + r + 8 * hi;
      Y1[m * RM_LD + n] = (bf16)acc1[r];
      Y2[m * RM_LD + n] = (bf16)acc2[r];
    }
  }
  __syncthreads();

  // ---- Phase 5: Z1 = Y1·as2, Z2 = Y2·as1  (128x96, K=96) — stored transposed ----
  for (int tile = wave; tile < 48; tile += 8) {
    const int tm = tile / 6, tn = tile % 6;
    v16bf a1 = afrag(Y1, RM_LD, tm * 16, 0);
    v16bf a2 = afrag(Y2, RM_LD, tm * 16, 0);
    v16bf b1 = bfragT(as2t, AS_LD, tn * 16, 0);
    v16bf b2 = bfragT(as1t, AS_LD, tn * 16, 0);
    v8f acc1 = {}, acc2 = {};
#pragma unroll
    for (int k0 = 0; k0 < Vp; k0 += 32) {
      const v16bf ca1 = a1, ca2 = a2, cb1 = b1, cb2 = b2;
      if (k0 + 32 < Vp) {
        a1 = afrag(Y1, RM_LD, tm * 16, k0 + 32);
        a2 = afrag(Y2, RM_LD, tm * 16, k0 + 32);
        b1 = bfragT(as2t, AS_LD, tn * 16, k0 + 32);
        b2 = bfragT(as1t, AS_LD, tn * 16, k0 + 32);
      }
      acc1 = WMMA_BF16(acc1, ca1, cb1);
      acc2 = WMMA_BF16(acc2, ca2, cb2);
    }
    const int n = tn * 16 + nl;
#pragma unroll
    for (int r = 0; r < 8; r += 2) {   // packed pair stores (contiguous in m)
      const int m = tm * 16 + r + 8 * hi;
      st2l(Z1t + n * TR_LD + m, acc1[r], acc1[r + 1]);
      st2l(Z2t + n * TR_LD + m, acc2[r], acc2[r + 1]);
    }
  }
  __syncthreads();

  // ---- Phase 6: proj = BN(W·Z) (256x96, K=128) + deterministic mean reduce ----
  {
    float ps1 = 0.f, ps2 = 0.f;
    for (int tile = wave; tile < 96; tile += 8) {
      const int tm = tile / 6, tn = tile % 6;
      // sW/oW for this lane's 8 output rows: unconditional 16B loads
      const f32x4 sL = ld4f(sW + tm * 16 + 8 * hi);
      const f32x4 sH = ld4f(sW + tm * 16 + 8 * hi + 4);
      const f32x4 oL = ld4f(oW + tm * 16 + 8 * hi);
      const f32x4 oH = ld4f(oW + tm * 16 + 8 * hi + 4);
      v16bf a = afrag(ww, CIn, tm * 16, 0);          // global bf16 W_w (T x CI)
      v16bf b1 = bfragT(Z1t, TR_LD, tn * 16, 0);
      v16bf b2 = bfragT(Z2t, TR_LD, tn * 16, 0);
      v8f acc1 = {}, acc2 = {};
#pragma unroll
      for (int k0 = 0; k0 < CIn; k0 += 32) {
        const v16bf ca = a, cb1 = b1, cb2 = b2;
        if (k0 + 32 < CIn) {
          a = afrag(ww, CIn, tm * 16, k0 + 32);
          b1 = bfragT(Z1t, TR_LD, tn * 16, k0 + 32);
          b2 = bfragT(Z2t, TR_LD, tn * 16, k0 + 32);
        }
        acc1 = WMMA_BF16(acc1, ca, cb1);
        acc2 = WMMA_BF16(acc2, ca, cb2);
      }
      const int n = tn * 16 + nl;
#pragma unroll
      for (int r = 0; r < 8; ++r) {
        const int t = tm * 16 + r + 8 * hi;
        const float sr = (r < 4) ? sL[r & 3] : sH[r & 3];
        const float orr = (r < 4) ? oL[r & 3] : oH[r & 3];
        const float p1 = acc1[r] * sr + orr;
        const float p2 = acc2[r] * sr + orr;
        P1[t * PR_LD + n] = (bf16)p1;
        P2[t * PR_LD + n] = (bf16)p2;
        if (n < Vn) { ps1 += p1; ps2 += p2; }
      }
    }
#pragma unroll
    for (int off = 16; off > 0; off >>= 1) {
      ps1 += __shfl_xor(ps1, off);
      ps2 += __shfl_xor(ps2, off);
    }
    if (lane == 0) { scl[16 + wave] = ps1; scl[24 + wave] = ps2; }
  }
  __syncthreads();

  // ---- Phase 7: 3-way scalar softmax blend per batch ----
  if (tid == 0) {
    float s1 = 0.f, s2 = 0.f;
    for (int wv = 0; wv < 8; ++wv) { s1 += scl[16 + wv]; s2 += scl[24 + wv]; }
    const float inv = 1.f / (float)TV;
    {
      const float a0 = scl[0] * inv, a1 = scl[1] * inv, a2 = s1 * inv;   // x1mean, cross_s2t mean (=x2mean), proj1 mean
      const float m = fmaxf(a0, fmaxf(a1, a2));
      const float e0 = __expf(a0 - m), e1 = __expf(a1 - m), e2 = __expf(a2 - m);
      const float si = 1.f / (e0 + e1 + e2);
      scl[8] = e0 * si; scl[9] = e1 * si; scl[10] = e2 * si;
    }
    {
      const float b0 = scl[1] * inv, b1 = scl[0] * inv, b2 = s2 * inv;   // x2mean, cross_t2s mean (=x1mean), proj2 mean
      const float m = fmaxf(b0, fmaxf(b1, b2));
      const float e0 = __expf(b0 - m), e1 = __expf(b1 - m), e2 = __expf(b2 - m);
      const float si = 1.f / (e0 + e1 + e2);
      scl[11] = e0 * si; scl[12] = e1 * si; scl[13] = e2 * si;
    }
  }
  __syncthreads();
  const float w10 = scl[8], w11 = scl[9], w12 = scl[10];
  const float w20 = scl[11], w21 = scl[12], w22 = scl[13];
  float* out2 = out + (size_t)NB * TV;
  {
    int t = tid / Vn, v = tid - (tid / Vn) * Vn;
    for (int i = tid; i < TV; i += 256) {
      const float xv1 = x1[gbase + i], xv2 = x2[gbase + i];
      out[gbase + i]  = w10 * xv1 + w11 * (x2mt[v] * (1.f / (float)Tn)) + w12 * (float)P1[t * PR_LD + v];
      out2[gbase + i] = w20 * xv2 + w21 * (x1mv[t] * (1.f / (float)Vn)) + w22 * (float)P2[t * PR_LD + v];
      v += 76; t += 2; if (v >= Vn) { v -= Vn; t += 1; }
    }
  }
}

// ---------------- host launcher ----------------
extern "C" void kernel_launch(void* const* d_in, const int* in_sizes, int n_in,
                              void* d_out, int out_size, void* d_ws, size_t ws_size,
                              hipStream_t stream) {
  (void)in_sizes; (void)n_in; (void)out_size; (void)ws_size;
  const float* x1     = (const float*)d_in[0];
  const float* x2     = (const float*)d_in[1];
  const float* g_bng  = (const float*)d_in[2];
  const float* g_bnb  = (const float*)d_in[3];
  const float* g_bnm  = (const float*)d_in[4];
  const float* g_bnv  = (const float*)d_in[5];
  const float* g_w    = (const float*)d_in[6];
  const float* g_b    = (const float*)d_in[7];
  const float* th_bng = (const float*)d_in[8];
  const float* th_bnb = (const float*)d_in[9];
  const float* th_bnm = (const float*)d_in[10];
  const float* th_bnv = (const float*)d_in[11];
  const float* th_w   = (const float*)d_in[12];
  const float* th_b   = (const float*)d_in[13];
  const float* ph_bng = (const float*)d_in[14];
  const float* ph_bnb = (const float*)d_in[15];
  const float* ph_bnm = (const float*)d_in[16];
  const float* ph_bnv = (const float*)d_in[17];
  const float* ph_w   = (const float*)d_in[18];
  const float* ph_b   = (const float*)d_in[19];
  const float* W_w    = (const float*)d_in[20];
  const float* W_b    = (const float*)d_in[21];
  const float* W_bng  = (const float*)d_in[22];
  const float* W_bnb  = (const float*)d_in[23];
  const float* W_bnm  = (const float*)d_in[24];
  const float* W_bnv  = (const float*)d_in[25];
  // d_in[26..33] (qt/ks/qs/kt): softmax over a singleton is identically 1 -> unused.

  char* ws = (char*)d_ws;
  bf16*  gw_bf  = (bf16*)(ws + WS_GW);
  bf16*  thw_bf = (bf16*)(ws + WS_THW);
  bf16*  phw_bf = (bf16*)(ws + WS_PHW);
  bf16*  ww_bf  = (bf16*)(ws + WS_WW);
  float* gb_e   = (float*)(ws + WS_GB);
  float* thb_e  = (float*)(ws + WS_THB);
  float* phb_e  = (float*)(ws + WS_PHB);
  float* sW     = (float*)(ws + WS_SW);
  float* oW     = (float*)(ws + WS_OW);

  prep_w_kernel<<<512, 256, 0, stream>>>(g_w, g_bng, g_bnv, th_w, th_bng, th_bnv,
                                         ph_w, ph_bng, ph_bnv, W_w,
                                         gw_bf, thw_bf, phw_bf, ww_bf);
  prep_b_kernel<<<1, 256, 0, stream>>>(g_w, g_bng, g_bnb, g_bnm, g_bnv, g_b,
                                       th_w, th_bng, th_bnb, th_bnm, th_bnv, th_b,
                                       ph_w, ph_bng, ph_bnb, ph_bnm, ph_bnv, ph_b,
                                       W_b, W_bng, W_bnb, W_bnm, W_bnv,
                                       gb_e, thb_e, phb_e, sW, oW);
  fused_attn_kernel<<<NB, 256, LDS_BYTES, stream>>>(x1, x2, gw_bf, thw_bf, phw_bf, ww_bf,
                                                    gb_e, thb_e, phb_e, sW, oW, (float*)d_out);
}
